// SAGEEncoder_48876727828948
// MI455X (gfx1250) — matrix operand order repaired and checked by reading, not compiled, not run on hardware
//
#include <hip/hip_runtime.h>

// ---------------------------------------------------------------------------
// GraphSAGE 2-layer encoder for MI455X (gfx1250, wave32, WMMA bf16)
//   layer: relu( (agg/deg) @ W_l + b + X @ W_r )
// Scatter is L2-resident atomic-add bound; GEMMs use v_wmma_f32_16x16x32_bf16.
// ---------------------------------------------------------------------------

#define DFEAT 128  // aggregation feature dim (both layers)

typedef __bf16 v16bf __attribute__((ext_vector_type(16)));
typedef float  v8f   __attribute__((ext_vector_type(8)));

__device__ __forceinline__ unsigned short f2bf(float f) {
  union { float f; unsigned u; } v; v.f = f;
  unsigned r = v.u + 0x7FFFu + ((v.u >> 16) & 1u);  // round-to-nearest-even
  return (unsigned short)(r >> 16);
}

__global__ void zero_kernel(float* __restrict__ p, long n) {
  long i = (long)blockIdx.x * blockDim.x + threadIdx.x;
  long stride = (long)gridDim.x * blockDim.x;
  for (; i < n; i += stride) p[i] = 0.0f;
}

__global__ void degree_kernel(const int* __restrict__ dst, float* __restrict__ deg, int n_edges) {
  int e = blockIdx.x * blockDim.x + threadIdx.x;
  if (e < n_edges) unsafeAtomicAdd(&deg[dst[e]], 1.0f);
}

// One wave per edge: 32 lanes x float4 = 128 features. Coalesced gather of
// x[src], hardware f32 atomic scatter-add into agg[dst] (L2-resident).
__global__ void scatter_kernel(const float* __restrict__ x, const int* __restrict__ src,
                               const int* __restrict__ dst, float* __restrict__ agg,
                               int n_edges) {
  long tid = (long)blockIdx.x * blockDim.x + threadIdx.x;
  long total = (long)n_edges * 32;
  if (tid >= total) return;
  int e = (int)(tid >> 5);
  int c = ((int)tid & 31) << 2;
  int s = src[e], d = dst[e];
  const float4 v = *(const float4*)(x + (long)s * DFEAT + c);
  float* o = agg + (long)d * DFEAT + c;
  unsafeAtomicAdd(o + 0, v.x);
  unsafeAtomicAdd(o + 1, v.y);
  unsafeAtomicAdd(o + 2, v.z);
  unsafeAtomicAdd(o + 3, v.w);
}

// Pre-pack f32 weight [K=128, N=dout] into WMMA-ready bf16 B-fragments:
// P[((ct*4 + kk)*32 + lane)*16 + j] = bf16( W[(kk*32 + (j>>3)*16 + (lane>>4)*8 + (j&7))*dout + ct*16 + (lane&15)] )
__global__ void pack_w_kernel(const float* __restrict__ W, unsigned short* __restrict__ P, int dout) {
  int t = blockIdx.x * blockDim.x + threadIdx.x;
  int total = (dout >> 4) * 128;  // tiles * 4 ksteps * 32 lanes
  if (t >= total) return;
  int lane = t & 31;
  int kk = (t >> 5) & 3;
  int ct = t >> 7;
  int col = lane & 15, hi = lane >> 4;
  int n = ct * 16 + col;
  unsigned short* o = P + t * 16;
#pragma unroll
  for (int j = 0; j < 16; ++j) {
    int k = kk * 32 + (j >> 3) * 16 + hi * 8 + (j & 7);
    o[j] = f2bf(W[k * dout + n]);
  }
}

// Fused: out = relu( (agg*1/deg) @ Wl + X @ Wr + bias ), K = 128.
// One block = 16 node rows; wave ct owns output columns [16ct, 16ct+16).
template <int DOUT>
__global__ void __launch_bounds__((DOUT / 16) * 32)
sage_gemm_kernel(const float* __restrict__ agg, const float* __restrict__ deg,
                 const float* __restrict__ xin, const unsigned short* __restrict__ Wl,
                 const unsigned short* __restrict__ Wr, const float* __restrict__ bias,
                 float* __restrict__ out, int n_nodes) {
  constexpr int NTHR = (DOUT / 16) * 32;
  constexpr int LDK = 136;  // pad 128 -> 136 (16B) for conflict-free ds_read_b128
  __shared__ __align__(16) unsigned short Am[16 * LDK];  // mean tile (bf16 bits)
  __shared__ __align__(16) unsigned short Ax[16 * LDK];  // x tile    (bf16 bits)
  __shared__ float rdeg[16];

  const int tid = threadIdx.x;
  const int node0 = blockIdx.x * 16;

  if (tid < 16) {
    int row = node0 + tid;
    float dg = (row < n_nodes) ? deg[row] : 1.0f;
    rdeg[tid] = 1.0f / fmaxf(dg, 1.0f);
  }
  __syncthreads();

  for (int i = tid; i < 16 * 128; i += NTHR) {
    int r = i >> 7, k = i & 127;
    int row = node0 + r;
    float m = 0.0f, xv = 0.0f;
    if (row < n_nodes) {
      long off = (long)row * 128 + k;
      m = agg[off] * rdeg[r];
      xv = xin[off];
    }
    Am[r * LDK + k] = f2bf(m);
    Ax[r * LDK + k] = f2bf(xv);
  }
  __syncthreads();

  const int wave = tid >> 5, lane = tid & 31;
  const int col = lane & 15, hi = lane >> 4;

  v8f acc;
  {
    float bv = bias[wave * 16 + col];
#pragma unroll
    for (int r = 0; r < 8; ++r) acc[r] = bv;
  }

#pragma unroll
  for (int kk = 0; kk < 4; ++kk) {
    union { uint4 q[2]; v16bf v; } am, ax, bl, br;
    // A frag (16-bit A 16x32 layout): lane row = lane&15, hi half selects K octet
    const unsigned short* pa = &Am[col * LDK + kk * 32 + hi * 8];
    am.q[0] = *(const uint4*)pa;          // K = kk*32 + hi*8 + [0..7]
    am.q[1] = *(const uint4*)(pa + 16);   // K = kk*32 + 16 + hi*8 + [0..7]
    const unsigned short* px = &Ax[col * LDK + kk * 32 + hi * 8];
    ax.q[0] = *(const uint4*)px;
    ax.q[1] = *(const uint4*)(px + 16);
    // B frags: pre-packed, 32 contiguous bytes per lane
    const unsigned short* pbl = Wl + ((wave * 4 + kk) * 32 + lane) * 16;
    bl.q[0] = *(const uint4*)pbl;
    bl.q[1] = *(const uint4*)(pbl + 8);
    const unsigned short* pbr = Wr + ((wave * 4 + kk) * 32 + lane) * 16;
    br.q[0] = *(const uint4*)pbr;
    br.q[1] = *(const uint4*)(pbr + 8);

    acc = __builtin_amdgcn_wmma_f32_16x16x32_bf16(false, am.v, false, bl.v,
                                                  (short)0, acc, false, false);
    acc = __builtin_amdgcn_wmma_f32_16x16x32_bf16(false, ax.v, false, br.v,
                                                  (short)0, acc, false, false);
  }

  // C/D layout: VGPR r -> M = r + 8*hi, N = lane&15
#pragma unroll
  for (int r = 0; r < 8; ++r) {
    int row = node0 + hi * 8 + r;
    if (row < n_nodes) {
      out[(long)row * DOUT + wave * 16 + col] = fmaxf(acc[r], 0.0f);
    }
  }
}

extern "C" void kernel_launch(void* const* d_in, const int* in_sizes, int n_in,
                              void* d_out, int out_size, void* d_ws, size_t ws_size,
                              hipStream_t stream) {
  (void)n_in; (void)out_size; (void)ws_size;
  const float* x   = (const float*)d_in[0];
  const int*   ei  = (const int*)d_in[1];
  const float* W1l = (const float*)d_in[2];
  const float* W1r = (const float*)d_in[3];
  const float* b1  = (const float*)d_in[4];
  const float* W2l = (const float*)d_in[5];
  const float* W2r = (const float*)d_in[6];
  const float* b2  = (const float*)d_in[7];
  float* out = (float*)d_out;

  const int n_nodes = in_sizes[0] / DFEAT;
  const int n_edges = in_sizes[1] / 2;
  const int* src = ei;
  const int* dst = ei + n_edges;

  char* ws = (char*)d_ws;
  size_t off = 0;
  auto alloc = [&](size_t bytes) -> void* {
    void* p = ws + off;
    off = (off + bytes + 255) & ~(size_t)255;
    return p;
  };
  float* agg = (float*)alloc((size_t)n_nodes * DFEAT * sizeof(float));
  float* h   = (float*)alloc((size_t)n_nodes * DFEAT * sizeof(float));
  float* deg = (float*)alloc((size_t)n_nodes * sizeof(float));
  unsigned short* pw1l = (unsigned short*)alloc(128 * 128 * 2);
  unsigned short* pw1r = (unsigned short*)alloc(128 * 128 * 2);
  unsigned short* pw2l = (unsigned short*)alloc(128 * 64 * 2);
  unsigned short* pw2r = (unsigned short*)alloc(128 * 64 * 2);

  const long agg_elems = (long)n_nodes * DFEAT;
  const int zb = 2048;  // blocks for grid-stride zero

  // ---- init + one-time prep ----
  zero_kernel<<<zb, 256, 0, stream>>>(agg, agg_elems);
  zero_kernel<<<zb, 256, 0, stream>>>(deg, (long)n_nodes);
  degree_kernel<<<(n_edges + 255) / 256, 256, 0, stream>>>(dst, deg, n_edges);
  pack_w_kernel<<<(8 * 128 + 255) / 256, 256, 0, stream>>>(W1l, pw1l, 128);
  pack_w_kernel<<<(8 * 128 + 255) / 256, 256, 0, stream>>>(W1r, pw1r, 128);
  pack_w_kernel<<<(4 * 128 + 255) / 256, 256, 0, stream>>>(W2l, pw2l, 64);
  pack_w_kernel<<<(4 * 128 + 255) / 256, 256, 0, stream>>>(W2r, pw2r, 64);

  const long sc_threads = (long)n_edges * 32;
  const int sc_blocks = (int)((sc_threads + 255) / 256);
  const int gemm_blocks = (n_nodes + 15) / 16;

  // ---- layer 1 ----
  scatter_kernel<<<sc_blocks, 256, 0, stream>>>(x, src, dst, agg, n_edges);
  sage_gemm_kernel<128><<<gemm_blocks, 256, 0, stream>>>(agg, deg, x, pw1l, pw1r, b1, h, n_nodes);

  // ---- layer 2 ----
  zero_kernel<<<zb, 256, 0, stream>>>(agg, agg_elems);
  scatter_kernel<<<sc_blocks, 256, 0, stream>>>(h, src, dst, agg, n_edges);
  sage_gemm_kernel<64><<<gemm_blocks, 128, 0, stream>>>(agg, deg, h, pw2l, pw2r, b2, out, n_nodes);
}
